// ScNet_21895743275449
// MI455X (gfx1250) — matrix operand order repaired and checked by reading, not compiled, not run on hardware
//
#include <hip/hip_runtime.h>
#include <math.h>

// ---------------------------------------------------------------------------
// ScNet (IL=False) for MI455X / gfx1250, restructured to be single-pass over
// the two 512MB tensors:
//   BB_sum = BB.sum(-1)                       (one pass over BB)
//   G_b    = B_b B_b^T  (128x4096 Gram)       (one pass over B, bf16x3 WMMA,
//                                              software-pipelined global->LDS)
//   c_b    = B_b . z_b                        (one pass over B)
//   per layer: aux matvec on BB_sum, BatchNorm, clip, loss via
//   dis = (||z||^2 - 2 x.c + x^T G x)/32768
// BB_sum (16MB) and G (16MB) stay resident in the 192MB L2 across the
// 8-layer loop, so per-layer traffic is effectively L2-bound.
// ---------------------------------------------------------------------------

#define BATCH 256
#define NN 128
#define KSUB 32
#define JLEN (NN * KSUB)   // 4096
#define NUM_LAYER 8

typedef __attribute__((ext_vector_type(16))) __bf16 v16bf;
typedef __attribute__((ext_vector_type(8)))  float  v8f;
typedef __attribute__((ext_vector_type(8)))  unsigned short ushort8_t;

union BfFrag {
    v16bf       bv;
    ushort8_t   h[2];
    unsigned short us[16];
};
static_assert(sizeof(BfFrag) == 32, "frag size");

// Workspace layout (float offsets). Requires ws_size >= ~34 MB.
static const size_t OFF_BBSUM = 0;                       // 256*128*128
static const size_t OFF_G     = OFF_BBSUM + (size_t)BATCH * NN * NN;
static const size_t OFF_BZ    = OFF_G     + (size_t)BATCH * NN * NN;  // 256*128
static const size_t OFF_C     = OFF_BZ    + (size_t)BATCH * NN;       // 256*128
static const size_t OFF_OUT   = OFF_C     + (size_t)BATCH * NN;       // 256*128
static const size_t OFF_MU    = OFF_OUT   + (size_t)BATCH * NN;       // 128
static const size_t OFF_RINV  = OFF_MU    + NN;                       // 128
static const size_t OFF_PA    = OFF_RINV  + NN;                       // 8*256
static const size_t OFF_PB    = OFF_PA    + NUM_LAYER * BATCH;        // 8*256
static const size_t OFF_Z2P   = OFF_PB    + NUM_LAYER * BATCH;        // 256
static const size_t OFF_Z2    = OFF_Z2P   + BATCH;                    // 1

__device__ __forceinline__ void bf16split(float f, unsigned short& hi, unsigned short& lo) {
    unsigned u = __float_as_uint(f);
    unsigned r = u + 0x7FFFu + ((u >> 16) & 1u);            // RNE to bf16
    hi = (unsigned short)(r >> 16);
    float fh = __uint_as_float((unsigned)hi << 16);
    float fl = f - fh;                                       // exact in f32
    unsigned u2 = __float_as_uint(fl);
    unsigned r2 = u2 + 0x7FFFu + ((u2 >> 16) & 1u);
    lo = (unsigned short)(r2 >> 16);
}

// --- sum over trailing K=32 axis: used for BB -> BB_sum and zB -> Bz_sum ----
__global__ __launch_bounds__(256) void reduce_k_kernel(const float* __restrict__ src,
                                                       float* __restrict__ dst,
                                                       long long total) {
    long long idx = (long long)blockIdx.x * 256 + threadIdx.x;
    if (idx >= total) return;
    const float4* p = (const float4*)(src + idx * KSUB);
    float s = 0.f;
#pragma unroll
    for (int q = 0; q < KSUB / 4; ++q) {
        float4 v = p[q];
        s += v.x + v.y + v.z + v.w;
    }
    dst[idx] = s;
}

// --- ||z||^2 partial sums ---------------------------------------------------
__global__ __launch_bounds__(256) void z2_partial_kernel(const float* __restrict__ z,
                                                         float* __restrict__ part) {
    __shared__ float red[256];
    const int t = threadIdx.x;
    const long long tid = (long long)blockIdx.x * 256 + t;
    const float4* z4 = (const float4*)z;
    float s = 0.f;
#pragma unroll
    for (int q = 0; q < 4; ++q) {
        float4 v = z4[tid + (long long)65536 * q];
        s += v.x * v.x + v.y * v.y + v.z * v.z + v.w * v.w;
    }
    red[t] = s; __syncthreads();
    for (int off = 128; off > 0; off >>= 1) {
        if (t < off) red[t] += red[t + off];
        __syncthreads();
    }
    if (t == 0) part[blockIdx.x] = red[0];
}

__global__ __launch_bounds__(256) void reduce256_kernel(const float* __restrict__ part,
                                                        float* __restrict__ out) {
    __shared__ float red[256];
    const int t = threadIdx.x;
    red[t] = part[t]; __syncthreads();
    for (int off = 128; off > 0; off >>= 1) {
        if (t < off) red[t] += red[t + off];
        __syncthreads();
    }
    if (t == 0) out[0] = red[0];
}

// --- c[b,n] = sum_{m,k} z[b,m,k] * B[b,n,m,k] ------------------------------
__global__ __launch_bounds__(256) void c_kernel(const float* __restrict__ Bmat,
                                                const float* __restrict__ z,
                                                float* __restrict__ c) {
    const int b = blockIdx.x;
    const int t = threadIdx.x;
    const int wave = t >> 5, lane = t & 31;
    __shared__ float4 zbuf[JLEN / 4];                        // 16 KB
    const float4* zsrc = (const float4*)(z + (size_t)b * JLEN);
#pragma unroll
    for (int q = 0; q < 4; ++q) zbuf[t + 256 * q] = zsrc[t + 256 * q];
    __syncthreads();
    for (int i = 0; i < 16; ++i) {
        const int n = wave * 16 + i;
        const float4* Brow = (const float4*)(Bmat + ((size_t)b * NN + n) * JLEN);
        float s = 0.f;
#pragma unroll 4
        for (int it = 0; it < 32; ++it) {
            float4 a = zbuf[it * 32 + lane];
            float4 bb = Brow[it * 32 + lane];
            s += a.x * bb.x + a.y * bb.y + a.z * bb.z + a.w * bb.w;
        }
#pragma unroll
        for (int off = 16; off > 0; off >>= 1) s += __shfl_xor(s, off, 32);
        if (lane == 0) c[(size_t)b * NN + n] = s;
    }
}

// --- Gram: G[b] = M M^T, M = B[b] as [128 x 4096], bf16 hi/lo x3 WMMA ------
// Software-pipelined: global loads for step ks+1 issue right after the
// produce-barrier and complete under the WMMA block of step ks.
__global__ __launch_bounds__(256) void gram_wmma_kernel(const float* __restrict__ Bmat,
                                                        float* __restrict__ G) {
    const int b    = blockIdx.x;
    const int t    = threadIdx.x;
    const int wave = t >> 5;
    const int lane = t & 31;
    const int g    = lane >> 4;     // half-wave select per ISA fragment layout
    const int lm   = lane & 15;

    __shared__ alignas(16) unsigned short sh_hi[NN * 32];   // 8 KB
    __shared__ alignas(16) unsigned short sh_lo[NN * 32];   // 8 KB
    unsigned int* sh_hi32 = (unsigned int*)sh_hi;
    unsigned int* sh_lo32 = (unsigned int*)sh_lo;

    const float* Mb = Bmat + (size_t)b * NN * JLEN;
    const int r  = t >> 1;          // staging: row
    const int hh = t & 1;           // staging: which 16-float half of the chunk
    const float* row_base = Mb + (size_t)r * JLEN + hh * 16;

    v8f acc[8] = {};                // 8 column tiles * 8 f32 accum regs

    // prefetch chunk 0 into registers
    float4 pre[4];
    {
        const float4* src4 = (const float4*)row_base;
#pragma unroll
        for (int q = 0; q < 4; ++q) pre[q] = src4[q];
    }

    for (int ks = 0; ks < JLEN / 32; ++ks) {
        // ---- convert prefetched chunk to bf16 hi/lo and store to LDS ----
        {
            const int base32 = r * 16 + hh * 8;             // uint index in LDS
#pragma unroll
            for (int q = 0; q < 4; ++q) {
                float4 v = pre[q];
                unsigned short h0, l0, h1, l1, h2, l2, h3, l3;
                bf16split(v.x, h0, l0); bf16split(v.y, h1, l1);
                bf16split(v.z, h2, l2); bf16split(v.w, h3, l3);
                sh_hi32[base32 + q * 2 + 0] = (unsigned)h0 | ((unsigned)h1 << 16);
                sh_hi32[base32 + q * 2 + 1] = (unsigned)h2 | ((unsigned)h3 << 16);
                sh_lo32[base32 + q * 2 + 0] = (unsigned)l0 | ((unsigned)l1 << 16);
                sh_lo32[base32 + q * 2 + 1] = (unsigned)l2 | ((unsigned)l3 << 16);
            }
        }
        __syncthreads();

        // ---- issue global loads for the NEXT chunk (latency hidden by WMMA)
        if (ks + 1 < JLEN / 32) {
            const float4* src4 = (const float4*)(row_base + (ks + 1) * 32);
#pragma unroll
            for (int q = 0; q < 4; ++q) pre[q] = src4[q];
        }

        // ---- A fragment (16x32 bf16): rows 16*wave..+15, ISA A layout ----
        BfFrag ah, al;
        {
            const int arow = 16 * wave + lm;
            const int o0 = arow * 32 + 8 * g;        // j=0..7  -> K = 8g + j
            const int o1 = arow * 32 + 16 + 8 * g;   // j=8..15 -> K = 16 + 8g + (j-8)
            ah.h[0] = *(const ushort8_t*)(sh_hi + o0);
            ah.h[1] = *(const ushort8_t*)(sh_hi + o1);
            al.h[0] = *(const ushort8_t*)(sh_lo + o0);
            al.h[1] = *(const ushort8_t*)(sh_lo + o1);
        }

#pragma unroll
        for (int cb = 0; cb < 8; ++cb) {
            // B fragment (32x16 bf16) = transposed tile: lane -> column n,
            // element j -> K = 8g + j (j<8), 16 + 8g + (j-8) (j>=8), matching A.
            const int brow = 16 * cb + lm;
            const int o0 = brow * 32 + 8 * g;
            BfFrag bh, bl;
            bh.h[0] = *(const ushort8_t*)(sh_hi + o0);
            bh.h[1] = *(const ushort8_t*)(sh_hi + o0 + 16);
            bl.h[0] = *(const ushort8_t*)(sh_lo + o0);
            bl.h[1] = *(const ushort8_t*)(sh_lo + o0 + 16);

            acc[cb] = __builtin_amdgcn_wmma_f32_16x16x32_bf16(
                false, ah.bv, false, bh.bv, (short)0, acc[cb], false, false);
            acc[cb] = __builtin_amdgcn_wmma_f32_16x16x32_bf16(
                false, ah.bv, false, bl.bv, (short)0, acc[cb], false, false);
            acc[cb] = __builtin_amdgcn_wmma_f32_16x16x32_bf16(
                false, al.bv, false, bh.bv, (short)0, acc[cb], false, false);
        }
        __syncthreads();   // LDS WAR guard before next convert/store phase
    }

    // ---- write back: C/D layout — VGPR rr: M = rr + 8g, N = lane&15 ----
#pragma unroll
    for (int cb = 0; cb < 8; ++cb) {
#pragma unroll
        for (int rr = 0; rr < 8; ++rr) {
            const int nrow = 16 * wave + 8 * g + rr;
            const int mcol = 16 * cb + lm;
            G[((size_t)b * NN + nrow) * NN + mcol] = acc[cb][rr];
        }
    }
}

// --- per-layer: out = (x BB_sum - Bz)*w1 + b1 + x*w2 + b2 -------------------
__global__ __launch_bounds__(128) void layer_out_kernel(const float* __restrict__ xest,
                                                        const float* __restrict__ BBsum,
                                                        const float* __restrict__ Bz,
                                                        const float* __restrict__ w1,
                                                        const float* __restrict__ b1,
                                                        const float* __restrict__ w2,
                                                        const float* __restrict__ b2,
                                                        float* __restrict__ out, int l) {
    const int b = blockIdx.x;
    const int m = threadIdx.x;
    __shared__ float xs[NN];
    xs[m] = xest[(size_t)b * NN + m];
    __syncthreads();
    const float* BBb = BBsum + (size_t)b * NN * NN;
    float s = 0.f;
#pragma unroll 4
    for (int n = 0; n < NN; ++n) s += xs[n] * BBb[n * NN + m];
    const float aux = s - Bz[(size_t)b * NN + m];
    out[(size_t)b * NN + m] = aux * w1[l * NN + m] + b1[l * NN + m]
                            + xs[m] * w2[l * NN + m] + b2[l * NN + m];
}

// --- BatchNorm1d training-mode stats (biased var, two-pass) -----------------
__global__ __launch_bounds__(128) void bn_stats_kernel(const float* __restrict__ out,
                                                       float* __restrict__ mu,
                                                       float* __restrict__ rinv) {
    const int n = threadIdx.x;
    float s = 0.f;
    for (int b = 0; b < BATCH; ++b) s += out[b * NN + n];
    const float m = s * (1.0f / BATCH);
    float s2 = 0.f;
    for (int b = 0; b < BATCH; ++b) { float d = out[b * NN + n] - m; s2 += d * d; }
    mu[n]   = m;
    rinv[n] = rsqrtf(s2 * (1.0f / BATCH) + 1e-5f);
}

// --- BN apply + mask + piecewise clip (== clamp(2v, -1, 1)) -----------------
__global__ __launch_bounds__(256) void bn_apply_kernel(const float* __restrict__ out,
                                                       const float* __restrict__ mu,
                                                       const float* __restrict__ rinv,
                                                       const float* __restrict__ gamma,
                                                       const float* __restrict__ beta,
                                                       const float* __restrict__ Mask,
                                                       float* __restrict__ xest, int l) {
    const int idx = blockIdx.x * 256 + threadIdx.x;          // 32768 total
    const int n = idx & (NN - 1);
    float v = (out[idx] - mu[n]) * rinv[n] * gamma[l * NN + n] + beta[l * NN + n];
    v *= Mask[idx];
    xest[idx] = fminf(fmaxf(2.0f * v, -1.0f), 1.0f);
}

// --- per-batch loss partials: x^T G x and x . c -----------------------------
__global__ __launch_bounds__(128) void loss_kernel(const float* __restrict__ xest,
                                                   const float* __restrict__ G,
                                                   const float* __restrict__ c,
                                                   float* __restrict__ pA,
                                                   float* __restrict__ pB, int l) {
    const int b = blockIdx.x;
    const int m = threadIdx.x;
    __shared__ float xs[NN];
    __shared__ float rA[NN], rB[NN];
    xs[m] = xest[(size_t)b * NN + m];
    __syncthreads();
    const float* Gb = G + (size_t)b * NN * NN;
    float y = 0.f;
#pragma unroll 4
    for (int n = 0; n < NN; ++n) y += xs[n] * Gb[n * NN + m];
    rA[m] = xs[m] * y;
    rB[m] = xs[m] * c[(size_t)b * NN + m];
    __syncthreads();
    for (int off = 64; off > 0; off >>= 1) {
        if (m < off) { rA[m] += rA[m + off]; rB[m] += rB[m + off]; }
        __syncthreads();
    }
    if (m == 0) { pA[l * BATCH + b] = rA[0]; pB[l * BATCH + b] = rB[0]; }
}

// --- final loss: log(l+1) * (Z2 - 2*xc + xGx) / 32768 -----------------------
__global__ __launch_bounds__(256) void finalize_kernel(const float* __restrict__ pA,
                                                       const float* __restrict__ pB,
                                                       const float* __restrict__ Z2,
                                                       float* __restrict__ loss_out) {
    const int l = blockIdx.x;
    const int t = threadIdx.x;
    __shared__ float rA[256], rB[256];
    rA[t] = pA[l * BATCH + t];
    rB[t] = pB[l * BATCH + t];
    __syncthreads();
    for (int off = 128; off > 0; off >>= 1) {
        if (t < off) { rA[t] += rA[t + off]; rB[t] += rB[t + off]; }
        __syncthreads();
    }
    if (t == 0) {
        const float dis = (Z2[0] - 2.0f * rB[0] + rA[0]) * (1.0f / (BATCH * NN));
        loss_out[l] = logf((float)(l + 1)) * dis;
    }
}

__global__ void zero_kernel(float* __restrict__ p, int n) {
    const int i = blockIdx.x * 256 + threadIdx.x;
    if (i < n) p[i] = 0.f;
}

extern "C" void kernel_launch(void* const* d_in, const int* in_sizes, int n_in,
                              void* d_out, int out_size, void* d_ws, size_t ws_size,
                              hipStream_t stream) {
    const float* BB    = (const float*)d_in[0];
    const float* zB    = (const float*)d_in[1];
    /* d_in[2] = x : unused, x_est starts at zeros_like(x) */
    const float* z     = (const float*)d_in[3];
    const float* Bmat  = (const float*)d_in[4];
    const float* Mask  = (const float*)d_in[5];
    const float* w1    = (const float*)d_in[6];
    const float* b1    = (const float*)d_in[7];
    const float* w2    = (const float*)d_in[8];
    const float* b2    = (const float*)d_in[9];
    const float* gamma = (const float*)d_in[10];
    const float* beta  = (const float*)d_in[11];

    float* xest = (float*)d_out;              // [256,128] final x_est
    float* loss = xest + BATCH * NN;          // [8] losses

    float* ws    = (float*)d_ws;
    float* BBsum = ws + OFF_BBSUM;
    float* G     = ws + OFF_G;
    float* Bz    = ws + OFF_BZ;
    float* cbuf  = ws + OFF_C;
    float* outb  = ws + OFF_OUT;
    float* mu    = ws + OFF_MU;
    float* rinv  = ws + OFF_RINV;
    float* pA    = ws + OFF_PA;
    float* pB    = ws + OFF_PB;
    float* z2p   = ws + OFF_Z2P;
    float* Z2    = ws + OFF_Z2;

    // x_est := 0 (fresh every call; deterministic)
    zero_kernel<<<(BATCH * NN + 255) / 256, 256, 0, stream>>>(xest, BATCH * NN);

    // Single-pass precomputations over the 512MB tensors
    reduce_k_kernel<<<(BATCH * NN * NN) / 256, 256, 0, stream>>>(BB, BBsum, (long long)BATCH * NN * NN);
    reduce_k_kernel<<<(BATCH * NN) / 256, 256, 0, stream>>>(zB, Bz, (long long)BATCH * NN);
    z2_partial_kernel<<<256, 256, 0, stream>>>(z, z2p);
    reduce256_kernel<<<1, 256, 0, stream>>>(z2p, Z2);
    c_kernel<<<BATCH, 256, 0, stream>>>(Bmat, z, cbuf);
    gram_wmma_kernel<<<BATCH, 256, 0, stream>>>(Bmat, G);

    for (int l = 0; l < NUM_LAYER; ++l) {
        layer_out_kernel<<<BATCH, NN, 0, stream>>>(xest, BBsum, Bz, w1, b1, w2, b2, outb, l);
        bn_stats_kernel<<<1, NN, 0, stream>>>(outb, mu, rinv);
        bn_apply_kernel<<<(BATCH * NN) / 256, 256, 0, stream>>>(outb, mu, rinv, gamma, beta, Mask, xest, l);
        loss_kernel<<<BATCH, NN, 0, stream>>>(xest, G, cbuf, pA, pB, l);
    }
    finalize_kernel<<<NUM_LAYER, 256, 0, stream>>>(pA, pB, Z2, loss);
}
